// DeformConv_927712936310
// MI455X (gfx1250) — matrix operand order repaired and checked by reading, not compiled, not run on hardware
//
#include <hip/hip_runtime.h>
#include <hip/hip_bf16.h>
#include <math.h>

// ---------------------------------------------------------------------------
// Deformable conv (K=3, pad=1) + BN + ReLU for B=8, Ci=Co=256, H=W=64.
// Both the main contraction (M=256,K=2304,N=32768 ~38.7 GFLOP) and the
// offset/modulation conv (M=27->32,K=2304,N=32768 ~4.8 GFLOP) run on
// v_wmma_f32_16x16x32_bf16. x (32MB) is L2-resident (192MB L2), so the
// bilinear gathers are L2-served. dcn_main double-buffers its LDS panel so
// tap k+1's gather staging co-executes (VALU/VMEM) under tap k's WMMAs.
// ---------------------------------------------------------------------------

typedef __attribute__((ext_vector_type(16))) __bf16 v16bf;
typedef __attribute__((ext_vector_type(2)))  __bf16 v2bf;
typedef __attribute__((ext_vector_type(8)))  float  v8f;

#define BB   8
#define CI   256
#define CO   256
#define HH   64
#define WW   64
#define HW   4096      // 64*64
#define K2   9
#define NPIX (BB * HW) // 32768
#define PROW 272       // panel row stride in bf16 (32B-aligned rows)
#define PSZ  (16 * PROW)

// im2col tile for the offsets conv: 4 rows x 66 cols (1-col halo each side)
#define TCOLS 66
#define TROWS 4
#define NPOS  (TROWS * TCOLS)          // 264
#define OFF_LDS_BYTES (NPOS * PROW * 2) // 143,616 B (dynamic LDS, WGP has 320KB)

// ---------------------------------------------------------------------------
// A-fragment packing (16-bit A 16x32, wave32; this code's contiguous k order):
//   lane<16 : row M=lane,    K = t
//   lane>=16: row M=lane-16, K = t + 16
// Fragment order: [coTile][tap][ciChunk][lane 0..31][t 0..15]
// ---------------------------------------------------------------------------

// Kernel 1a: pack main weight[Co][Ci][3][3] fp32 -> bf16 A-fragments.
__global__ __launch_bounds__(256) void dcn_prepack(const float* __restrict__ wt,
                                                   __bf16* __restrict__ wp) {
    int i = blockIdx.x * 256 + threadIdx.x;            // 0 .. 589823
    if (i >= 16 * K2 * 8 * 512) return;
    int t      = i & 15;
    int lane   = (i >> 4) & 31;
    int frag   = i >> 9;                               // 0 .. 1151
    int chunk  = frag & 7;
    int k      = (frag >> 3) % K2;
    int coTile = frag / (K2 * 8);
    int M      = lane & 15;
    int K      = t + ((lane >= 16) ? 16 : 0);
    int co     = coTile * 16 + M;
    int ci     = chunk * 32 + K;
    wp[i] = (__bf16)wt[(co * CI + ci) * K2 + k];
}

// Kernel 1b: pack offset(18ch)+mod(9ch) weights, zero-padded to 32 rows.
__global__ __launch_bounds__(256) void dcn_prepack_off(const float* __restrict__ offw,
                                                       const float* __restrict__ modw,
                                                       __bf16* __restrict__ wp2) {
    int i = blockIdx.x * 256 + threadIdx.x;            // 0 .. 73727
    if (i >= 2 * K2 * 8 * 512) return;
    int t      = i & 15;
    int lane   = (i >> 4) & 31;
    int frag   = i >> 9;                               // 0 .. 143
    int chunk  = frag & 7;
    int k      = (frag >> 3) % K2;
    int coTile = frag / (K2 * 8);                      // 0..1
    int M      = lane & 15;
    int K      = t + ((lane >= 16) ? 16 : 0);
    int co     = coTile * 16 + M;                      // 0..31
    int ci     = chunk * 32 + K;
    float v = 0.f;
    if (co < 18)      v = offw[(co * CI + ci) * K2 + k];
    else if (co < 27) v = modw[((co - 18) * CI + ci) * K2 + k];
    wp2[i] = (__bf16)v;
}

// ---------------------------------------------------------------------------
// Kernel 2: offsets/modulation conv3x3 via WMMA.
// Block = 512 thr = 16 waves, owns 128 pixels (2 image rows).
// Stage zero-padded im2col tile [264 pos][256 ci] bf16 ONCE, then all 9 taps
// read it with shifted pointers: wave w = (ntile = w>>1, coTile = w&1),
// 72 WMMA per wave. Epilogue emits dy/dx and mask = 2*sigmoid.
// ---------------------------------------------------------------------------
__global__ __launch_bounds__(512) void dcn_offsets_wmma(const float* __restrict__ x,
                                                        const __bf16* __restrict__ wp2,
                                                        const float* __restrict__ offb,
                                                        const float* __restrict__ modb,
                                                        float* __restrict__ dy,
                                                        float* __restrict__ dxo,
                                                        float* __restrict__ msk) {
    extern __shared__ __attribute__((aligned(32))) char dynsm[];
    __bf16* tile = (__bf16*)dynsm;

    int tid = threadIdx.x;
    int p0  = blockIdx.x * 128;
    int b   = p0 >> 12;
    int hw0 = p0 & 4095;
    int h0  = hw0 >> 6;

    // ---- stage tile: pos = tilerow*66 + tilecol maps to image
    //      (h0-1+tilerow, tilecol-1); OOB -> 0. Each (pos, ci-range) once.
    {
        int pos, ci0, cin;
        if (tid < 248)      { pos = tid;       ci0 = 0;   cin = 128; }
        else if (tid < 264) { pos = tid;       ci0 = 0;   cin = 256; }
        else                { pos = tid - 264; ci0 = 128; cin = 128; }
        int trow = pos / TCOLS, tcol = pos - trow * TCOLS;
        int imr = h0 - 1 + trow;
        int imc = tcol - 1;
        bool valid = (imr >= 0) && (imr < HH) && (imc >= 0) && (imc < WW);
        const float* xbase = x + (((size_t)b) << 20);  // uniform (SGPR) base
        int boff = imr * WW + imc;                     // elements
        __bf16* trow_p = &tile[pos * PROW];
        for (int c = ci0; c < ci0 + cin; c += 2) {
            float s0 = 0.f, s1 = 0.f;
            if (valid) {
                int o = c * HW + boff;                 // 32-bit elem offset
                s0 = xbase[o];
                s1 = xbase[o + HW];
            }
            *(v2bf*)&trow_p[c] = (v2bf){(__bf16)s0, (__bf16)s1};
        }
    }
    __syncthreads();

    int wave   = tid >> 5;
    int lane   = tid & 31;
    int nb     = lane & 15;
    int half   = lane >> 4;
    int ntile  = wave >> 1;     // 0..7 -> pixels ntile*16 .. +15
    int coTile = wave & 1;
    int prow   = ntile >> 2;                 // image row within block (0/1)
    int pcol0  = (ntile & 3) * 16;           // starting col of the 16 pixels

    v8f acc = {};
#pragma unroll
    for (int k = 0; k < K2; ++k) {
        int ky = k / 3, kx = k % 3;
        int pos0 = (prow + ky) * TCOLS + (pcol0 + kx);
        const __bf16* bp  = &tile[(pos0 + nb) * PROW + half * 16];
        const __bf16* wpk = wp2 + ((size_t)((coTile * K2 + k) * 8)) * 512 + lane * 16;
#pragma unroll
        for (int c = 0; c < 8; ++c) {
            v16bf a  = *(const v16bf*)(wpk + c * 512);
            v16bf bf = *(const v16bf*)(bp + c * 32);
            acc = __builtin_amdgcn_wmma_f32_16x16x32_bf16(
                false, a, false, bf, (short)0, acc, false, false);
        }
    }

    // C/D layout: VGPR r, lane -> M = r + 8*half, N = nb.
#pragma unroll
    for (int r = 0; r < 8; ++r) {
        int co = coTile * 16 + half * 8 + r;
        int hw = hw0 + ntile * 16 + nb;
        if (co < 18) {
            int k = co >> 1;
            size_t oi = ((size_t)(b * K2 + k)) * HW + hw;
            float v = acc[r] + offb[co];
            if (co & 1) dxo[oi] = v; else dy[oi] = v;
        } else if (co < 27) {
            int k = co - 18;
            size_t oi = ((size_t)(b * K2 + k)) * HW + hw;
            float m = acc[r] + modb[k];
            msk[oi] = 2.f / (1.f + expf(-m));
        }
    }
}

// ---------------------------------------------------------------------------
// Kernel 3: fused gather + modulated GEMM. Block = 512 thr = 16 waves, owns
// 16 pixels; wave w owns Co tile w. Panel is DOUBLE-BUFFERED: while tap k's
// 8 WMMAs run on buffer k&1, tap k+1's bilinear gather stages buffer (k+1)&1
// (16-bit WMMA co-executes with VALU), with one barrier per tap.
// Staging uses hoisted corner tables + saddr-form loads (uniform base,
// 32-bit varying offset, per-chunk stride folded into the 24-bit IOFFSET).
// ---------------------------------------------------------------------------
__global__ __launch_bounds__(512) void dcn_main(const float* __restrict__ x,
                                                const float* __restrict__ dy,
                                                const float* __restrict__ dxo,
                                                const float* __restrict__ msk,
                                                const __bf16* __restrict__ wp,
                                                const float* __restrict__ bias,
                                                float* __restrict__ out) {
    __shared__ __attribute__((aligned(32))) __bf16 panel[2 * PSZ];
    __shared__ __attribute__((aligned(16))) float tw4[K2 * 16 * 4]; // corner wgts
    __shared__ __attribute__((aligned(16))) int   ti4[K2 * 16 * 4]; // corner byte offs

    int tid = threadIdx.x;
    int p0  = blockIdx.x << 4;
    int b   = p0 >> 12;
    int hw0 = p0 & 4095;

    // Per-(tap,pixel) bilinear setup: 144 tasks; offsets pre-scaled to bytes.
    if (tid < K2 * 16) {
        int k = tid >> 4, n = tid & 15;
        int hw = hw0 + n;
        int h = hw >> 6, wc = hw & 63;
        size_t oi = ((size_t)(b * K2 + k)) * HW + hw;
        float py = (float)(h - 1 + k / 3) + dy[oi];
        float px = (float)(wc - 1 + k % 3) + dxo[oi];
        float m  = msk[oi];
        float y0f = floorf(py), x0f = floorf(px);
        int   y0 = (int)y0f,    x0 = (int)x0f;
        float wy1 = py - y0f,   wx1 = px - x0f;
        float wy0 = 1.f - wy1,  wx0 = 1.f - wx1;
        int   y1 = y0 + 1,      x1 = x0 + 1;
        float vy0 = (y0 >= 0 && y0 < HH) ? 1.f : 0.f;
        float vy1 = (y1 >= 0 && y1 < HH) ? 1.f : 0.f;
        float vx0 = (x0 >= 0 && x0 < WW) ? 1.f : 0.f;
        float vx1 = (x1 >= 0 && x1 < WW) ? 1.f : 0.f;
        int cy0 = min(max(y0, 0), HH - 1), cy1 = min(max(y1, 0), HH - 1);
        int cx0 = min(max(x0, 0), WW - 1), cx1 = min(max(x1, 0), WW - 1);
        float* w4 = &tw4[tid * 4];
        int*   i4 = &ti4[tid * 4];
        w4[0] = wy0 * wx0 * vy0 * vx0 * m;
        w4[1] = wy0 * wx1 * vy0 * vx1 * m;
        w4[2] = wy1 * wx0 * vy1 * vx0 * m;
        w4[3] = wy1 * wx1 * vy1 * vx1 * m;
        i4[0] = (cy0 * WW + cx0) * 4; i4[1] = (cy0 * WW + cx1) * 4;
        i4[2] = (cy1 * WW + cx0) * 4; i4[3] = (cy1 * WW + cx1) * 4;
    }

    int wave = tid >> 5;      // Co tile
    int lane = tid & 31;
    int nb   = lane & 15;     // B-fragment column (pixel)
    int half = lane >> 4;
    int n4   = tid & 15;      // staged pixel (invariant: 512 % 16 == 0)
    int c0   = tid >> 4;      // staged ci base lane 0..31
    const char* xc = (const char*)(x + (((size_t)b) << 20)); // uniform base

    // Bilinear-stage tap k into LDS buffer `buf`.
    auto stage = [&](int k, __bf16* buf) {
        float4 w4 = *(const float4*)&tw4[(k * 16 + n4) * 4];
        int4   o4 = *(const int4*)&ti4[(k * 16 + n4) * 4];
        int cib = c0 << 14;   // ci byte offset (c0 * 4096 * 4)
        int oa = cib + o4.x, ob2 = cib + o4.y, oc = cib + o4.z, od = cib + o4.w;
        __bf16* prow_p = &buf[n4 * PROW + c0];
#pragma unroll
        for (int i = 0; i < 8; ++i) {
            const int C = i * (32 * HW * 4);  // < 2^23: folds into IOFFSET
            float s = w4.x * *(const float*)(xc + oa  + C)
                    + w4.y * *(const float*)(xc + ob2 + C)
                    + w4.z * *(const float*)(xc + oc  + C)
                    + w4.w * *(const float*)(xc + od  + C);
            prow_p[i * 32] = (__bf16)s;
        }
    };

    v8f acc = {};
    __syncthreads();          // corner tables visible
    stage(0, &panel[0]);
    __syncthreads();          // buffer 0 staged

    for (int k = 0; k < K2; ++k) {
        __bf16* cur = &panel[(k & 1) * PSZ];
        if (k + 1 < K2) stage(k + 1, &panel[((k + 1) & 1) * PSZ]);

        const __bf16* wpk = wp + ((size_t)((wave * K2 + k) * 8)) * 512 + lane * 16;
        const __bf16* bp  = &cur[nb * PROW + half * 16];
        if (k + 1 < K2) __builtin_prefetch(wpk + 8 * 512, 0, 1);  // next tap's A
#pragma unroll
        for (int c = 0; c < 8; ++c) {
            v16bf a  = *(const v16bf*)(wpk + c * 512);
            v16bf bf = *(const v16bf*)(bp + c * 32);
            acc = __builtin_amdgcn_wmma_f32_16x16x32_bf16(
                false, a, false, bf, (short)0, acc, false, false);
        }
        __syncthreads();      // next buffer staged / cur free for reuse
    }

    // C/D layout: VGPR r, lane -> M = r + 8*half, N = nb.
#pragma unroll
    for (int r = 0; r < 8; ++r) {
        int co = wave * 16 + half * 8 + r;
        out[((size_t)(b * CO + co)) * HW + hw0 + nb] = acc[r] + bias[co];
    }
}

// ---------------------------------------------------------------------------
// Kernel 4: per-channel mean / rstd over (B,H,W) = 32768 values.
// ---------------------------------------------------------------------------
__global__ __launch_bounds__(256) void dcn_stats(const float* __restrict__ out,
                                                 float* __restrict__ stats) {
    __shared__ float s1[256], s2[256];
    int ch = blockIdx.x;
    float sum = 0.f, sq = 0.f;
    for (int i = threadIdx.x; i < NPIX; i += 256) {
        int b = i >> 12, hw = i & 4095;
        float v = out[((size_t)(b * CO + ch)) * HW + hw];
        sum += v;
        sq  += v * v;
    }
    s1[threadIdx.x] = sum;
    s2[threadIdx.x] = sq;
    __syncthreads();
    for (int st = 128; st > 0; st >>= 1) {
        if (threadIdx.x < st) {
            s1[threadIdx.x] += s1[threadIdx.x + st];
            s2[threadIdx.x] += s2[threadIdx.x + st];
        }
        __syncthreads();
    }
    if (threadIdx.x == 0) {
        float mu  = s1[0] * (1.f / (float)NPIX);
        float var = s2[0] * (1.f / (float)NPIX) - mu * mu;
        stats[ch]      = mu;
        stats[CO + ch] = rsqrtf(var + 1e-5f);
    }
}

// ---------------------------------------------------------------------------
// Kernel 5: in-place normalize + affine + ReLU.
// ---------------------------------------------------------------------------
__global__ __launch_bounds__(256) void dcn_bnrelu(float* __restrict__ out,
                                                  const float* __restrict__ stats,
                                                  const float* __restrict__ gamma,
                                                  const float* __restrict__ beta) {
    size_t i = (size_t)blockIdx.x * 256 + threadIdx.x;
    int co = (int)((i >> 12) & (CO - 1));
    float v = out[i];
    float y = (v - stats[co]) * stats[CO + co] * gamma[co] + beta[co];
    out[i] = y > 0.f ? y : 0.f;
}

// ---------------------------------------------------------------------------
extern "C" void kernel_launch(void* const* d_in, const int* in_sizes, int n_in,
                              void* d_out, int out_size, void* d_ws, size_t ws_size,
                              hipStream_t stream) {
    (void)in_sizes; (void)n_in; (void)out_size; (void)ws_size;
    const float* x     = (const float*)d_in[0];
    const float* offw  = (const float*)d_in[1];
    const float* offb  = (const float*)d_in[2];
    const float* modw  = (const float*)d_in[3];
    const float* modb  = (const float*)d_in[4];
    const float* wt    = (const float*)d_in[5];
    const float* bias  = (const float*)d_in[6];
    const float* gamma = (const float*)d_in[7];
    const float* beta  = (const float*)d_in[8];
    float* out = (float*)d_out;

    // Workspace: dy | dx | mask | packed main W | packed off W | stats (~4.9 MB)
    const size_t NOFF = (size_t)BB * K2 * HW;            // 294912 floats
    float*  dy    = (float*)d_ws;
    float*  dxo   = dy + NOFF;
    float*  msk   = dxo + NOFF;
    __bf16* wp    = (__bf16*)(msk + NOFF);               // 589824 bf16
    __bf16* wp2   = wp + (size_t)16 * K2 * 8 * 512;      // 73728 bf16
    float*  stats = (float*)(wp2 + (size_t)2 * K2 * 8 * 512);

    dcn_prepack<<<(16 * K2 * 8 * 512 + 255) / 256, 256, 0, stream>>>(wt, wp);
    dcn_prepack_off<<<(2 * K2 * 8 * 512 + 255) / 256, 256, 0, stream>>>(offw, modw, wp2);
    dcn_offsets_wmma<<<NPIX / 128, 512, OFF_LDS_BYTES, stream>>>(
        x, wp2, offb, modb, dy, dxo, msk);
    dcn_main<<<NPIX / 16, 512, 0, stream>>>(x, dy, dxo, msk, wp, bias, out);
    dcn_stats<<<CO, 256, 0, stream>>>(out, stats);
    dcn_bnrelu<<<(int)(((size_t)BB * CO * HW) / 256), 256, 0, stream>>>(out, stats, gamma, beta);
}
